// ESIMPredModel_20220706030283
// MI455X (gfx1250) — compile-verified, matrix operand order checked by reading, not run on hardware
//
#include <hip/hip_runtime.h>
#include <cstddef>

// ---------------- problem constants ----------------
#define BB   128      // batch
#define LL   128      // sequence length
#define EMBD 300      // embedding dim (= H)
#define HH   300      // hidden
#define GG   1200     // 4*H gates
#define KXE  320      // emb K padded to mult of 32
#define KXC  2400     // comparator input K (8*300, already mult of 32)
#define KH   320      // hidden K padded
#define LD_BF 608     // padded 2H for bf16 activations (600 -> 608)
#define NEGV (-1e30f)

// ---------------- WMMA bf16 plumbing ----------------
typedef __attribute__((ext_vector_type(16))) __bf16   v16bf;
typedef __attribute__((ext_vector_type(8)))  float    v8f;
typedef __attribute__((ext_vector_type(4)))  unsigned uint4v;

union FragBF { v16bf v; uint4v q[2]; unsigned u[8]; };

__device__ __forceinline__ unsigned short f2bf(float f) {
  unsigned u = __float_as_uint(f);
  unsigned r = 0x7FFFu + ((u >> 16) & 1u);   // round-to-nearest-even
  return (unsigned short)((u + r) >> 16);
}

// A/B fragment from a 16B-aligned row-major row.
// ISA 7.12.2 16-bit 16x32 layout: VGPR i holds K pair k0 + (i/4)*16 + half*8 + (i%4)*2,
// i.e. VGPRs 0-3 = 8 consecutive bf16 at k0+half*8, VGPRs 4-7 = 8 at k0+16+half*8.
// Both groups are one global_load_b128 each.
__device__ __forceinline__ v16bf frag_row(const unsigned short* rowp, int k0, int half) {
  FragBF f;
  f.q[0] = *(const uint4v*)(rowp + k0 + (half << 3));
  f.q[1] = *(const uint4v*)(rowp + k0 + 16 + (half << 3));
  return f.v;
}

// B fragment when B is stored [K][N] row-major (strided 16-bit gathers).
__device__ __forceinline__ v16bf frag_b_kn(const unsigned short* Bp, int ldb, int col,
                                           int k0, int half) {
  FragBF f;
#pragma unroll
  for (int i = 0; i < 8; ++i) {
    int k = k0 + ((i >> 2) << 4) + (half << 3) + ((i & 3) << 1);
    unsigned lo = Bp[(size_t)k * ldb + col];
    unsigned hi = Bp[(size_t)(k + 1) * ldb + col];
    f.u[i] = lo | (hi << 16);
  }
  return f.v;
}

#define WMMA_BF16(a, b, c) \
  __builtin_amdgcn_wmma_f32_16x16x32_bf16(false, (a), false, (b), (short)0, (c), false, false)

__device__ __forceinline__ float sigm(float x) { return 1.0f / (1.0f + expf(-x)); }

// ---------------- elementwise / prep kernels ----------------

__global__ void k_embed(const int* __restrict__ x, const float* __restrict__ emb,
                        unsigned short* __restrict__ e) {
  long long idx = (long long)blockIdx.x * blockDim.x + threadIdx.x;
  long long total = 2LL * BB * LL * KXE;
  if (idx >= total) return;
  int k = (int)(idx % KXE);
  int t = (int)((idx / KXE) % LL);
  int b = (int)((idx / ((long long)KXE * LL)) % BB);
  int s = (int)(idx / ((long long)KXE * LL * BB));
  int tok = x[((size_t)b * 2 + s) * LL + t];
  float val = (k < EMBD) ? emb[(size_t)tok * EMBD + k] : 0.0f;
  e[idx] = f2bf(val);
}

__global__ void k_mask(const int* __restrict__ x, float* __restrict__ mask) {
  int idx = blockIdx.x * blockDim.x + threadIdx.x;
  if (idx >= 2 * BB * LL) return;
  int t = idx % LL;
  int b = (idx / LL) % BB;
  int s = idx / (LL * BB);
  int tok = x[((size_t)b * 2 + s) * LL + t];
  mask[((size_t)s * BB + b) * LL + t] = (tok != 0) ? 1.0f : 0.0f;
}

__global__ void k_len(const float* __restrict__ mask, int* __restrict__ len) {
  int idx = blockIdx.x * blockDim.x + threadIdx.x;
  if (idx >= 2 * BB) return;
  const float* m = mask + (size_t)idx * LL;
  int s = 0;
  for (int t = 0; t < LL; ++t) s += (m[t] > 0.0f) ? 1 : 0;
  len[idx] = s;
}

// Fused transposed bf16 weights: Wc[dir][n][k], k<KxPad from Wx (zero-padded),
// k in [KxPad, KxPad+320) from Wh (zero-padded).
__global__ void k_prep_w(const float* __restrict__ Wx, const float* __restrict__ Wh,
                         unsigned short* __restrict__ Wc, int KxReal, int KxPad) {
  long long KA = (long long)KxPad + KH;
  long long idx = (long long)blockIdx.x * blockDim.x + threadIdx.x;
  long long total = 2LL * GG * KA;
  if (idx >= total) return;
  int k = (int)(idx % KA);
  int n = (int)((idx / KA) % GG);
  int dir = (int)(idx / (KA * GG));
  float val;
  if (k < KxPad) {
    val = (k < KxReal) ? Wx[((size_t)dir * KxReal + k) * GG + n] : 0.0f;
  } else {
    int kk = k - KxPad;
    val = (kk < HH) ? Wh[((size_t)dir * HH + kk) * GG + n] : 0.0f;
  }
  Wc[idx] = f2bf(val);
}

__global__ void k_zero32(unsigned* __restrict__ p, long long n) {
  long long idx = (long long)blockIdx.x * blockDim.x + threadIdx.x;
  if (idx < n) p[idx] = 0u;
}

// ---------------- LSTM step: fused [x_t ; h] @ [Wx ; Wh] on WMMA ----------------
// KX is compile-time: two branch-free K loops (x-part, h-part).
// Each wave computes TWO 16x16 z tiles (adjacent batch tiles) sharing one weight
// (B) fragment: grid (GG/16, BB/32, 2 dirs), block 32.
template <int KX>
__global__ void k_lstm_gemm(const unsigned short* __restrict__ X,
                            const unsigned short* __restrict__ Wc,
                            const unsigned short* __restrict__ hbf,
                            float* __restrict__ z,
                            const int* __restrict__ len, int t) {
  constexpr int KA = KX + KH;
  int dir = blockIdx.z;
  int n0 = blockIdx.x << 4;
  int m0 = blockIdx.y << 5;
  int lane = threadIdx.x;
  int r = lane & 15, half = lane >> 4;
  int b0 = m0 + r, b1 = m0 + 16 + r;
  int ln0 = len[b0], ln1 = len[b1];
  int tp0 = (dir == 0) ? t : ((t < ln0) ? (ln0 - 1 - t) : t);
  int tp1 = (dir == 0) ? t : ((t < ln1) ? (ln1 - 1 - t) : t);
  const unsigned short* x0 = X + ((size_t)b0 * LL + tp0) * KX;
  const unsigned short* x1 = X + ((size_t)b1 * LL + tp1) * KX;
  const unsigned short* h0 = hbf + ((size_t)dir * BB + b0) * KH;
  const unsigned short* h1 = hbf + ((size_t)dir * BB + b1) * KH;
  const unsigned short* wrow = Wc + ((size_t)dir * GG + (n0 + r)) * (size_t)KA;
  v8f acc0 = {}, acc1 = {};
#pragma unroll 2
  for (int k0 = 0; k0 < KX; k0 += 32) {
    __builtin_prefetch(wrow + k0 + 32, 0, 3);
    v16bf wb = frag_row(wrow, k0, half);
    v16bf a0 = frag_row(x0, k0, half);
    v16bf a1 = frag_row(x1, k0, half);
    acc0 = WMMA_BF16(a0, wb, acc0);
    acc1 = WMMA_BF16(a1, wb, acc1);
  }
#pragma unroll 2
  for (int k0 = 0; k0 < KH; k0 += 32) {
    v16bf wb = frag_row(wrow, KX + k0, half);
    v16bf a0 = frag_row(h0, k0, half);
    v16bf a1 = frag_row(h1, k0, half);
    acc0 = WMMA_BF16(a0, wb, acc0);
    acc1 = WMMA_BF16(a1, wb, acc1);
  }
  float* zb = z + (size_t)dir * BB * GG;
#pragma unroll
  for (int i = 0; i < 8; ++i) {
    zb[(size_t)(m0 + half * 8 + i) * GG + (n0 + r)] = acc0[i];
    zb[(size_t)(m0 + 16 + half * 8 + i) * GG + (n0 + r)] = acc1[i];
  }
}

// Gate nonlinearity + masked state update + output scatter (handles fwd/bwd concat).
__global__ void k_gate(const float* __restrict__ z, const float* __restrict__ bias,
                       float* __restrict__ hf, float* __restrict__ cf,
                       unsigned short* __restrict__ hbf,
                       const float* __restrict__ mask, const int* __restrict__ len,
                       float* __restrict__ yf, unsigned short* __restrict__ ybf, int t) {
  int idx = blockIdx.x * blockDim.x + threadIdx.x;
  if (idx >= 2 * BB * HH) return;
  int j = idx % HH;
  int b = (idx / HH) % BB;
  int dir = idx / (HH * BB);
  int ln = len[b];
  int tpos = (dir == 0) ? t : ((t < ln) ? (ln - 1 - t) : t);
  const float* zr = z + ((size_t)dir * BB + b) * GG;
  const float* br = bias + (size_t)dir * GG;
  float zi = zr[j] + br[j];
  float zfg = zr[HH + j] + br[HH + j];
  float zg = zr[2 * HH + j] + br[2 * HH + j];
  float zo = zr[3 * HH + j] + br[3 * HH + j];
  size_t st = ((size_t)dir * BB + b) * KH + j;
  float c_old = cf[st], h_old = hf[st];
  float cn = sigm(zfg) * c_old + sigm(zi) * tanhf(zg);
  float hn = sigm(zo) * tanhf(cn);
  float mt = mask[(size_t)b * LL + tpos];
  float c = mt * cn + (1.0f - mt) * c_old;
  float h = mt * hn + (1.0f - mt) * h_old;
  cf[st] = c;
  hf[st] = h;
  hbf[st] = f2bf(h);
  float yv = mt * hn;
  yf[((size_t)b * LL + tpos) * (2 * HH) + (size_t)dir * HH + j] = yv;
  if (ybf) ybf[((size_t)b * LL + tpos) * LD_BF + (size_t)dir * HH + j] = f2bf(yv);
}

// ---------------- attention GEMMs on WMMA ----------------
// scores[b] = prem_bf[b] (128x608) @ hyp_bf[b]^T.
// Two i-tiles per wave share the hyp (B) fragment: grid (8, 4, 128), block 32.
__global__ void k_scores(const unsigned short* __restrict__ pbf,
                         const unsigned short* __restrict__ hbfm,
                         float* __restrict__ scores) {
  int bz = blockIdx.z;
  int j0 = blockIdx.x << 4;
  int i0 = blockIdx.y << 5;
  int lane = threadIdx.x;
  int r = lane & 15, half = lane >> 4;
  const unsigned short* a0row = pbf + ((size_t)bz * LL + (i0 + r)) * LD_BF;
  const unsigned short* a1row = pbf + ((size_t)bz * LL + (i0 + 16 + r)) * LD_BF;
  const unsigned short* brow  = hbfm + ((size_t)bz * LL + (j0 + r)) * LD_BF;
  v8f acc0 = {}, acc1 = {};
#pragma unroll 2
  for (int k0 = 0; k0 < LD_BF; k0 += 32) {
    v16bf bm = frag_row(brow, k0, half);
    v16bf a0 = frag_row(a0row, k0, half);
    v16bf a1 = frag_row(a1row, k0, half);
    acc0 = WMMA_BF16(a0, bm, acc0);
    acc1 = WMMA_BF16(a1, bm, acc1);
  }
#pragma unroll
  for (int i = 0; i < 8; ++i) {
    scores[((size_t)bz * LL + (i0 + half * 8 + i)) * LL + (j0 + r)] = acc0[i];
    scores[((size_t)bz * LL + (i0 + 16 + half * 8 + i)) * LL + (j0 + r)] = acc1[i];
  }
}

// C[b] (128x608 f32) = A[b] (128x128 bf16 row-major) @ Bkn[b] (128x608 bf16 [K][N]).
// Two m-tiles per wave share the strided B fragment: grid (38, 4, 128), block 32.
__global__ void k_attn_gemm(const unsigned short* __restrict__ A,
                            const unsigned short* __restrict__ Bkn,
                            float* __restrict__ C) {
  int bz = blockIdx.z;
  int n0 = blockIdx.x << 4;
  int m0 = blockIdx.y << 5;
  int lane = threadIdx.x;
  int r = lane & 15, half = lane >> 4;
  const unsigned short* a0row = A + (size_t)bz * LL * LL + (size_t)(m0 + r) * LL;
  const unsigned short* a1row = A + (size_t)bz * LL * LL + (size_t)(m0 + 16 + r) * LL;
  const unsigned short* Bb = Bkn + (size_t)bz * LL * LD_BF;
  int col = n0 + r;
  v8f acc0 = {}, acc1 = {};
#pragma unroll
  for (int k0 = 0; k0 < LL; k0 += 32) {
    v16bf bm = frag_b_kn(Bb, LD_BF, col, k0, half);
    v16bf a0 = frag_row(a0row, k0, half);
    v16bf a1 = frag_row(a1row, k0, half);
    acc0 = WMMA_BF16(a0, bm, acc0);
    acc1 = WMMA_BF16(a1, bm, acc1);
  }
#pragma unroll
  for (int i = 0; i < 8; ++i) {
    C[((size_t)bz * LL + (m0 + half * 8 + i)) * LD_BF + col] = acc0[i];
    C[((size_t)bz * LL + (m0 + 16 + half * 8 + i)) * LD_BF + col] = acc1[i];
  }
}

// softmax over j with mask_h; emit alpha bf16 [b][i][j]
__global__ void k_softmax_alpha(const float* __restrict__ scores,
                                const float* __restrict__ maskh,
                                unsigned short* __restrict__ alpha) {
  int idx = blockIdx.x * blockDim.x + threadIdx.x;
  if (idx >= BB * LL) return;
  int i = idx % LL, b = idx / LL;
  const float* srow = scores + ((size_t)b * LL + i) * LL;
  const float* mrow = maskh + (size_t)b * LL;
  float mx = -3.0e38f;
  for (int j = 0; j < LL; ++j) {
    float s = (mrow[j] > 0.0f) ? srow[j] : NEGV;
    mx = fmaxf(mx, s);
  }
  float sum = 0.0f;
  for (int j = 0; j < LL; ++j) {
    float s = (mrow[j] > 0.0f) ? srow[j] : NEGV;
    sum += expf(s - mx);
  }
  float inv = 1.0f / sum;
  unsigned short* arow = alpha + ((size_t)b * LL + i) * LL;
  for (int j = 0; j < LL; ++j) {
    float s = (mrow[j] > 0.0f) ? srow[j] : NEGV;
    arow[j] = f2bf(expf(s - mx) * inv);
  }
}

// softmax over i with mask_p; emit transposed betaT bf16 [b][j][i]
__global__ void k_softmax_beta(const float* __restrict__ scores,
                               const float* __restrict__ maskp,
                               unsigned short* __restrict__ betaT) {
  int idx = blockIdx.x * blockDim.x + threadIdx.x;
  if (idx >= BB * LL) return;
  int j = idx % LL, b = idx / LL;
  const float* mrow = maskp + (size_t)b * LL;
  float mx = -3.0e38f;
  for (int i = 0; i < LL; ++i) {
    float s = (mrow[i] > 0.0f) ? scores[((size_t)b * LL + i) * LL + j] : NEGV;
    mx = fmaxf(mx, s);
  }
  float sum = 0.0f;
  for (int i = 0; i < LL; ++i) {
    float s = (mrow[i] > 0.0f) ? scores[((size_t)b * LL + i) * LL + j] : NEGV;
    sum += expf(s - mx);
  }
  float inv = 1.0f / sum;
  unsigned short* brow = betaT + ((size_t)b * LL + j) * LL;
  for (int i = 0; i < LL; ++i) {
    float s = (mrow[i] > 0.0f) ? scores[((size_t)b * LL + i) * LL + j] : NEGV;
    brow[i] = f2bf(expf(s - mx) * inv);
  }
}

// m[b][t][2400] bf16 = concat(bi, attn, bi-attn, bi*attn)
__global__ void k_build_m(const float* __restrict__ bi, const float* __restrict__ attn,
                          unsigned short* __restrict__ m) {
  long long idx = (long long)blockIdx.x * blockDim.x + threadIdx.x;
  long long total = (long long)BB * LL * (2 * HH);
  if (idx >= total) return;
  int d = (int)(idx % (2 * HH));
  long long bt = idx / (2 * HH);
  float x1 = bi[bt * (2 * HH) + d];
  float x2 = attn[bt * LD_BF + d];
  unsigned short* row = m + bt * KXC;
  row[d] = f2bf(x1);
  row[600 + d] = f2bf(x2);
  row[1200 + d] = f2bf(x1 - x2);
  row[1800 + d] = f2bf(x1 * x2);
}

// pooling: v = [v1_ave | v2_ave | v1_max | v2_max], each 600 wide
__global__ void k_pool(const float* __restrict__ v1f, const float* __restrict__ v2f,
                       const int* __restrict__ lenp, const int* __restrict__ lenh,
                       float* __restrict__ v) {
  int idx = blockIdx.x * blockDim.x + threadIdx.x;
  if (idx >= BB * (2 * HH)) return;
  int d = idx % (2 * HH);
  int b = idx / (2 * HH);
  float s1 = 0.0f, m1 = -3.0e38f, s2 = 0.0f, m2 = -3.0e38f;
  for (int t = 0; t < LL; ++t) {
    float a = v1f[((size_t)b * LL + t) * (2 * HH) + d];
    float c = v2f[((size_t)b * LL + t) * (2 * HH) + d];
    s1 += a; m1 = fmaxf(m1, a);
    s2 += c; m2 = fmaxf(m2, c);
  }
  float* vr = v + (size_t)b * 2400;
  vr[d] = s1 / (float)lenp[b];
  vr[600 + d] = s2 / (float)lenh[b];
  vr[1200 + d] = m1;
  vr[1800 + d] = m2;
}

__global__ void k_mlp(const float* __restrict__ v, const float* __restrict__ Wm,
                      const float* __restrict__ bm, float* __restrict__ hm) {
  int idx = blockIdx.x * blockDim.x + threadIdx.x;
  if (idx >= BB * EMBD) return;
  int o = idx % EMBD;
  int b = idx / EMBD;
  const float* vr = v + (size_t)b * 2400;
  float s = bm[o];
  for (int k = 0; k < 2400; ++k) s += vr[k] * Wm[(size_t)k * EMBD + o];
  hm[(size_t)b * EMBD + o] = tanhf(s);
}

__global__ void k_cls(const float* __restrict__ hm, const float* __restrict__ Wc,
                      const float* __restrict__ bc, float* __restrict__ out) {
  int idx = blockIdx.x * blockDim.x + threadIdx.x;
  if (idx >= BB * 3) return;
  int o = idx % 3;
  int b = idx / 3;
  float s = bc[o];
  for (int k = 0; k < EMBD; ++k) s += hm[(size_t)b * EMBD + k] * Wc[k * 3 + o];
  out[(size_t)b * 3 + o] = s;
}

// ---------------- host orchestration ----------------
static inline dim3 launch1d(long long total) {
  return dim3((unsigned)((total + 255) / 256));
}

static inline void zero_dev(void* p, size_t bytes, hipStream_t stream) {
  long long n = (long long)(bytes / 4);
  k_zero32<<<launch1d(n), 256, 0, stream>>>((unsigned*)p, n);
}

template <int KX>
static void run_bilstm(const unsigned short* X, const unsigned short* Wc,
                       const float* bias, const float* maskS, const int* lenS,
                       float* yf, unsigned short* ybf,
                       float* zbuf, float* hf, float* cfst, unsigned short* hbst,
                       hipStream_t stream) {
  zero_dev(hf,   2ull * BB * KH * 4, stream);
  zero_dev(cfst, 2ull * BB * KH * 4, stream);
  zero_dev(hbst, 2ull * BB * KH * 2, stream);
  for (int t = 0; t < LL; ++t) {
    k_lstm_gemm<KX><<<dim3(GG / 16, BB / 32, 2), dim3(32), 0, stream>>>(
        X, Wc, hbst, zbuf, lenS, t);
    k_gate<<<launch1d(2LL * BB * HH), 256, 0, stream>>>(
        zbuf, bias, hf, cfst, hbst, maskS, lenS, yf, ybf, t);
  }
}

extern "C" void kernel_launch(void* const* d_in, const int* in_sizes, int n_in,
                              void* d_out, int out_size, void* d_ws, size_t ws_size,
                              hipStream_t stream) {
  (void)in_sizes; (void)n_in; (void)out_size; (void)ws_size;
  const int*   x        = (const int*)d_in[0];
  const float* emb      = (const float*)d_in[1];
  const float* encp_Wx  = (const float*)d_in[2];
  const float* encp_Wh  = (const float*)d_in[3];
  const float* encp_b   = (const float*)d_in[4];
  const float* ench_Wx  = (const float*)d_in[5];
  const float* ench_Wh  = (const float*)d_in[6];
  const float* ench_b   = (const float*)d_in[7];
  const float* cmpp_Wx  = (const float*)d_in[8];
  const float* cmpp_Wh  = (const float*)d_in[9];
  const float* cmpp_b   = (const float*)d_in[10];
  const float* cmph_Wx  = (const float*)d_in[11];
  const float* cmph_Wh  = (const float*)d_in[12];
  const float* cmph_b   = (const float*)d_in[13];
  const float* W_mlp    = (const float*)d_in[14];
  const float* b_mlp    = (const float*)d_in[15];
  const float* W_cl     = (const float*)d_in[16];
  const float* b_cl     = (const float*)d_in[17];
  float* out = (float*)d_out;

  char* base = (char*)d_ws;
  size_t off = 0;
  auto alloc = [&](size_t bytes) -> void* {
    void* p = base + off;
    off = (off + bytes + 255) & ~(size_t)255;
    return p;
  };

  unsigned short* e       = (unsigned short*)alloc(2ull * BB * LL * KXE * 2);
  float*          maskA   = (float*)alloc(2ull * BB * LL * 4);
  int*            lenA    = (int*)alloc(2ull * BB * 4);
  unsigned short* WcEncP  = (unsigned short*)alloc(2ull * GG * (KXE + KH) * 2);
  unsigned short* WcEncH  = (unsigned short*)alloc(2ull * GG * (KXE + KH) * 2);
  unsigned short* WcCmpP  = (unsigned short*)alloc(2ull * GG * (KXC + KH) * 2);
  unsigned short* WcCmpH  = (unsigned short*)alloc(2ull * GG * (KXC + KH) * 2);
  float*          zbuf    = (float*)alloc(2ull * BB * GG * 4);
  float*          hf      = (float*)alloc(2ull * BB * KH * 4);
  float*          cfst    = (float*)alloc(2ull * BB * KH * 4);
  unsigned short* hbst    = (unsigned short*)alloc(2ull * BB * KH * 2);
  float*          premF   = (float*)alloc((size_t)BB * LL * (2 * HH) * 4);
  float*          hypF    = (float*)alloc((size_t)BB * LL * (2 * HH) * 4);
  unsigned short* premBf  = (unsigned short*)alloc((size_t)BB * LL * LD_BF * 2);
  unsigned short* hypBf   = (unsigned short*)alloc((size_t)BB * LL * LD_BF * 2);
  float*          scores  = (float*)alloc((size_t)BB * LL * LL * 4);
  unsigned short* alphaBf = (unsigned short*)alloc((size_t)BB * LL * LL * 2);
  unsigned short* betaTBf = (unsigned short*)alloc((size_t)BB * LL * LL * 2);
  float*          pAttn   = (float*)alloc((size_t)BB * LL * LD_BF * 4);
  float*          hAttn   = (float*)alloc((size_t)BB * LL * LD_BF * 4);
  unsigned short* mBuf    = (unsigned short*)alloc((size_t)BB * LL * KXC * 2);
  float*          v1F     = (float*)alloc((size_t)BB * LL * (2 * HH) * 4);
  float*          v2F     = (float*)alloc((size_t)BB * LL * (2 * HH) * 4);
  float*          vPool   = (float*)alloc((size_t)BB * 2400 * 4);
  float*          hMlp    = (float*)alloc((size_t)BB * EMBD * 4);

  const float* maskP = maskA;
  const float* maskH = maskA + (size_t)BB * LL;
  const int*   lenP  = lenA;
  const int*   lenH  = lenA + BB;
  const unsigned short* eP = e;
  const unsigned short* eH = e + (size_t)BB * LL * KXE;

  // ---- embed / mask / length ----
  k_embed<<<launch1d(2LL * BB * LL * KXE), 256, 0, stream>>>(x, emb, e);
  k_mask<<<launch1d(2LL * BB * LL), 256, 0, stream>>>(x, maskA);
  k_len<<<1, 256, 0, stream>>>(maskA, lenA);

  // ---- weight prep (fused [Wx;Wh], transposed, bf16) ----
  k_prep_w<<<launch1d(2LL * GG * (KXE + KH)), 256, 0, stream>>>(encp_Wx, encp_Wh, WcEncP, EMBD, KXE);
  k_prep_w<<<launch1d(2LL * GG * (KXE + KH)), 256, 0, stream>>>(ench_Wx, ench_Wh, WcEncH, EMBD, KXE);
  k_prep_w<<<launch1d(2LL * GG * (KXC + KH)), 256, 0, stream>>>(cmpp_Wx, cmpp_Wh, WcCmpP, KXC, KXC);
  k_prep_w<<<launch1d(2LL * GG * (KXC + KH)), 256, 0, stream>>>(cmph_Wx, cmph_Wh, WcCmpH, KXC, KXC);

  zero_dev(premBf, (size_t)BB * LL * LD_BF * 2, stream);   // pad cols stay zero
  zero_dev(hypBf,  (size_t)BB * LL * LD_BF * 2, stream);

  // ---- encoder BiLSTMs ----
  run_bilstm<KXE>(eP, WcEncP, encp_b, maskP, lenP, premF, premBf,
                  zbuf, hf, cfst, hbst, stream);
  run_bilstm<KXE>(eH, WcEncH, ench_b, maskH, lenH, hypF, hypBf,
                  zbuf, hf, cfst, hbst, stream);

  // ---- attention ----
  k_scores<<<dim3(LL / 16, LL / 32, BB), dim3(32), 0, stream>>>(premBf, hypBf, scores);
  k_softmax_alpha<<<launch1d((long long)BB * LL), 256, 0, stream>>>(scores, maskH, alphaBf);
  k_softmax_beta<<<launch1d((long long)BB * LL), 256, 0, stream>>>(scores, maskP, betaTBf);
  k_attn_gemm<<<dim3(LD_BF / 16, LL / 32, BB), dim3(32), 0, stream>>>(alphaBf, hypBf, pAttn);
  k_attn_gemm<<<dim3(LD_BF / 16, LL / 32, BB), dim3(32), 0, stream>>>(betaTBf, premBf, hAttn);

  // ---- comparator BiLSTMs (reuse mBuf for m_a then m_b) ----
  k_build_m<<<launch1d((long long)BB * LL * (2 * HH)), 256, 0, stream>>>(premF, pAttn, mBuf);
  run_bilstm<KXC>(mBuf, WcCmpP, cmpp_b, maskP, lenP, v1F, nullptr,
                  zbuf, hf, cfst, hbst, stream);
  k_build_m<<<launch1d((long long)BB * LL * (2 * HH)), 256, 0, stream>>>(hypF, hAttn, mBuf);
  run_bilstm<KXC>(mBuf, WcCmpH, cmph_b, maskH, lenH, v2F, nullptr,
                  zbuf, hf, cfst, hbst, stream);

  // ---- pooling + MLP + classifier ----
  k_pool<<<launch1d((long long)BB * (2 * HH)), 256, 0, stream>>>(v1F, v2F, lenP, lenH, vPool);
  k_mlp<<<launch1d((long long)BB * EMBD), 256, 0, stream>>>(vPool, W_mlp, b_mlp, hMlp);
  k_cls<<<launch1d((long long)BB * 3), 256, 0, stream>>>(hMlp, W_cl, b_cl, out);
}
